// NGCF_8804682957333
// MI455X (gfx1250) — matrix operand order repaired and checked by reading, not compiled, not run on hardware
//
#include <hip/hip_runtime.h>
#include <hip/hip_bf16.h>

// NGCF on MI455X (gfx1250, wave32). All heavy GEMMs use V_WMMA_F32_16X16X32_BF16:
// operands converted f32->bf16 in registers, f32 accumulation. The dominant cost
// is streaming adj (604 MB over 2 layers) at HBM speed (~26us floor at 23.3TB/s);
// bf16 WMMA makes the matrix math negligible against that. side_gemm uses a
// double-buffered LDS tile (one barrier per K-step) so global loads stay in
// flight under the WMMAs, plus a branch-free global_prefetch two tiles ahead.

typedef __attribute__((ext_vector_type(16))) __bf16 v16bf;
typedef __attribute__((ext_vector_type(8)))  float  v8f;

#define NB      8
#define NU      1024
#define NI      2048
#define NODES   3072
#define DIM     64
#define NLAYER  2
#define EPITCH  192   // (NLAYER+1)*DIM

// ---------------------------------------------------------------------------
// Convert layer weights f32 -> bf16 (row-major, 2 x 64 x 64 each)
// ---------------------------------------------------------------------------
__global__ void wconvert_kernel(const float* __restrict__ gc_w,
                                const float* __restrict__ bi_w,
                                __bf16* __restrict__ gcw16,
                                __bf16* __restrict__ biw16) {
  int i = blockIdx.x * blockDim.x + threadIdx.x;
  if (i < NLAYER * DIM * DIM) {
    gcw16[i] = (__bf16)gc_w[i];
    biw16[i] = (__bf16)bi_w[i];
  }
}

// ---------------------------------------------------------------------------
// Gather embeddings: all_e[b][node][0:64] = table row (f32), and transposed
// bf16 copy egoT[b][d][node] used as WMMA B-operand of the adj GEMM.
// grid (3072,1,8) x 64 threads
// ---------------------------------------------------------------------------
__global__ void gather_kernel(const int* __restrict__ uidx,
                              const int* __restrict__ iidx,
                              const float* __restrict__ utab,
                              const float* __restrict__ itab,
                              float* __restrict__ all_e,
                              __bf16* __restrict__ egoT) {
  const int b = blockIdx.z;
  const int node = blockIdx.x;
  const int d = threadIdx.x;
  float v;
  if (node < NU) v = utab[(size_t)uidx[b * NU + node] * DIM + d];
  else           v = itab[(size_t)iidx[b * NI + (node - NU)] * DIM + d];
  all_e[((size_t)b * NODES + node) * EPITCH + d] = v;
  egoT[((size_t)b * DIM + d) * NODES + node] = (__bf16)v;
}

// ---------------------------------------------------------------------------
// side[b] = adj[b] (3072x3072 f32) @ ego[b] (3072x64), bf16 WMMA, f32 accum.
// Block = 128 threads (4 waves), 64x64 output tile; wave w -> rows [16w,16w+16).
// adj K-tile (64x32 f32) staged coalesced -> bf16 LDS, double-buffered.
// grid (48,1,8)
// ---------------------------------------------------------------------------
__global__ void side_gemm_kernel(const float* __restrict__ adj,
                                 const __bf16* __restrict__ egoT,
                                 float* __restrict__ side) {
  __shared__ __bf16 aTile[2][64 * 32];       // 2 x 4 KB

  const int b     = blockIdx.z;
  const int mBase = blockIdx.x * 64;
  const int tid   = threadIdx.x;
  const int lane  = tid & 31;
  const int wave  = tid >> 5;                // 0..3
  const int half  = (lane < 16) ? 1 : 0;     // 1 = lanes 0-15
  const int l15   = lane & 15;

  const float*  adjB  = adj  + (size_t)b * NODES * NODES;
  const __bf16* egoTB = egoT + (size_t)b * DIM * NODES;

  v8f acc[4] = {};                            // 4 N-tiles of 16

  // coalesced load of one 64x32 f32 tile -> bf16 LDS buffer
  auto load_tile = [&](int buf, int kk) {
#pragma unroll
    for (int it = 0; it < 4; ++it) {
      int idx = (tid + it * 128) * 4;        // 0..2044
      int r = idx >> 5, c = idx & 31;
      const float4 v = *(const float4*)(adjB + (size_t)(mBase + r) * NODES + kk + c);
      __bf16* dst = &aTile[buf][r * 32 + c];
      dst[0] = (__bf16)v.x; dst[1] = (__bf16)v.y;
      dst[2] = (__bf16)v.z; dst[3] = (__bf16)v.w;
    }
  };

  load_tile(0, 0);
  __syncthreads();

  for (int kk = 0; kk < NODES; kk += 32) {
    const int cur = (kk >> 5) & 1;

    // branch-free prefetch two K-tiles ahead (one 128B line per tile row;
    // 128 lanes map onto 64 rows, duplicates coalesce in cache)
    {
      const int kpf = (kk + 64 <= NODES - 32) ? (kk + 64) : (NODES - 32);
      __builtin_prefetch(adjB + (size_t)(mBase + (tid & 63)) * NODES + kpf, 0, 2);
    }

    // stage next tile into the other buffer (uniform scalar branch)
    if (kk + 32 < NODES) load_tile(cur ^ 1, kk + 32);

    // A fragment (16x32 bf16): lanes 0-15 K {0..15}; lanes 16-31 K {16..31}
    v16bf a;
    {
      const __bf16* base = &aTile[cur][(wave * 16 + l15) * 32];
      const int c0 = half ? 0 : 16;
#pragma unroll
      for (int i = 0; i < 8; ++i) { a[i] = base[c0 + i]; a[8 + i] = base[c0 + 8 + i]; }
    }

    // B fragments from transposed ego: 16 contiguous K per lane
#pragma unroll
    for (int nt = 0; nt < 4; ++nt) {
      const __bf16* bsrc = egoTB + (size_t)(nt * 16 + l15) * NODES + kk + (half ? 0 : 16);
      v16bf bf;
#pragma unroll
      for (int i = 0; i < 16; ++i) bf[i] = bsrc[i];
      acc[nt] = __builtin_amdgcn_wmma_f32_16x16x32_bf16(false, a, false, bf,
                                                        (short)0, acc[nt], false, false);
    }
    __syncthreads();   // cur reads done + next-buffer writes visible
  }

  // store: VGPR r -> M = r + (lanes>=16 ? 8 : 0), N = lane&15
  float* outB = side + (size_t)b * NODES * DIM;
#pragma unroll
  for (int nt = 0; nt < 4; ++nt) {
#pragma unroll
    for (int r = 0; r < 8; ++r) {
      int m = mBase + wave * 16 + r + (half ? 0 : 8);
      int n = nt * 16 + l15;
      outB[(size_t)m * DIM + n] = acc[nt][r];
    }
  }
}

// ---------------------------------------------------------------------------
// Per-layer epilogue:
//   s  = leaky( side @ gc_w.T + gc_b )
//   bi = leaky( (ego*side) @ bi_w.T + bi_b )
//   ego_next = s + bi -> all_e col block (l+1)*64, and transposed bf16 egoT.
// B(k,n) = W[n][k] -> K-contiguous rows of W, no transpose needed.
// grid (48,1,8) x 128 threads
// ---------------------------------------------------------------------------
__global__ void combine_kernel(const float* __restrict__ side,
                               float* __restrict__ all_e,
                               __bf16* __restrict__ egoT,
                               const __bf16* __restrict__ gcw,   // 64x64 layer slice
                               const __bf16* __restrict__ biw,
                               const float* __restrict__ gcb,    // 64
                               const float* __restrict__ bib,
                               int lcur) {
  const int b    = blockIdx.z;
  const int tid  = threadIdx.x;
  const int lane = tid & 31;
  const int wave = tid >> 5;
  const int half = (lane < 16) ? 1 : 0;
  const int l15  = lane & 15;
  const int mRow = blockIdx.x * 64 + wave * 16 + l15;

  const float* sp = side  + ((size_t)b * NODES + mRow) * DIM;
  const float* ep = all_e + ((size_t)b * NODES + mRow) * EPITCH + lcur * DIM;

  v8f accS[4] = {}; v8f accB[4] = {};

  for (int kk = 0; kk < DIM; kk += 32) {
    v16bf aS, aB;
    const int c0 = kk + (half ? 0 : 16);
#pragma unroll
    for (int i = 0; i < 8; ++i) {
      float s0 = sp[c0 + i],      s1 = sp[c0 + 8 + i];
      float e0 = ep[c0 + i],      e1 = ep[c0 + 8 + i];
      aS[i] = (__bf16)s0;         aS[8 + i] = (__bf16)s1;
      aB[i] = (__bf16)(s0 * e0);  aB[8 + i] = (__bf16)(s1 * e1);
    }
#pragma unroll
    for (int nt = 0; nt < 4; ++nt) {
      const int n = nt * 16 + l15;
      const __bf16* gp = gcw + (size_t)n * DIM + kk + (half ? 0 : 16);
      const __bf16* bp = biw + (size_t)n * DIM + kk + (half ? 0 : 16);
      v16bf bg, bb;
#pragma unroll
      for (int i = 0; i < 16; ++i) { bg[i] = gp[i]; bb[i] = bp[i]; }
      accS[nt] = __builtin_amdgcn_wmma_f32_16x16x32_bf16(false, aS, false, bg,
                                                         (short)0, accS[nt], false, false);
      accB[nt] = __builtin_amdgcn_wmma_f32_16x16x32_bf16(false, aB, false, bb,
                                                         (short)0, accB[nt], false, false);
    }
  }

  float*  outE = all_e + (size_t)b * NODES * EPITCH;
  __bf16* egT  = egoT  + (size_t)b * DIM * NODES;
  const int mBaseW = blockIdx.x * 64 + wave * 16 + (half ? 0 : 8);
#pragma unroll
  for (int nt = 0; nt < 4; ++nt) {
    const int n = nt * 16 + l15;
    const float gb = gcb[n], bbv = bib[n];
#pragma unroll
    for (int r = 0; r < 8; ++r) {
      const int m = mBaseW + r;
      float s  = accS[nt][r] + gb;  s  = (s  > 0.f) ? s  : 0.01f * s;
      float bi = accB[nt][r] + bbv; bi = (bi > 0.f) ? bi : 0.01f * bi;
      const float e = s + bi;
      outE[(size_t)m * EPITCH + (lcur + 1) * DIM + n] = e;
      egT[(size_t)n * NODES + m] = (__bf16)e;
    }
  }
}

// ---------------------------------------------------------------------------
// scores[b] = u_g (1024x192) @ i_g.T (192x2048), bf16 WMMA, written to d_out.
// Block = 256 threads (8 waves): wave -> 16x64 sub-tile of the 64x128 WG tile.
// grid (16,16,8)
// ---------------------------------------------------------------------------
__global__ void scores_gemm_kernel(const float* __restrict__ all_e,
                                   float* __restrict__ out) {
  const int b    = blockIdx.z;
  const int tid  = threadIdx.x;
  const int lane = tid & 31;
  const int wave = tid >> 5;                 // 0..7
  const int half = (lane < 16) ? 1 : 0;
  const int l15  = lane & 15;
  const int mBase = blockIdx.y * 64 + (wave >> 1) * 16;
  const int nBase = blockIdx.x * 128 + (wave & 1) * 64;

  const float* E    = all_e + (size_t)b * NODES * EPITCH;
  const float* urow = E + (size_t)(mBase + l15) * EPITCH;

  v8f acc[4] = {};
  for (int kk = 0; kk < EPITCH; kk += 32) {
    v16bf a;
    const int c0 = kk + (half ? 0 : 16);
#pragma unroll
    for (int i = 0; i < 8; ++i) { a[i] = (__bf16)urow[c0 + i]; a[8 + i] = (__bf16)urow[c0 + 8 + i]; }
#pragma unroll
    for (int nt = 0; nt < 4; ++nt) {
      const int n = nBase + nt * 16 + l15;
      const float* irow = E + (size_t)(NU + n) * EPITCH + kk + (half ? 0 : 16);
      v16bf bf;
#pragma unroll
      for (int i = 0; i < 16; ++i) bf[i] = (__bf16)irow[i];
      acc[nt] = __builtin_amdgcn_wmma_f32_16x16x32_bf16(false, a, false, bf,
                                                        (short)0, acc[nt], false, false);
    }
  }

  float* O = out + (size_t)b * NU * NI;
#pragma unroll
  for (int nt = 0; nt < 4; ++nt) {
    const int n = nBase + nt * 16 + l15;
#pragma unroll
    for (int r = 0; r < 8; ++r) {
      const int m = mBase + r + (half ? 0 : 8);
      O[(size_t)m * NI + n] = acc[nt][r];
    }
  }
}

// ---------------------------------------------------------------------------
// In-place row-wise log-softmax over 2048 columns. grid = 8*1024 rows.
// ---------------------------------------------------------------------------
__global__ void logsoftmax_kernel(float* __restrict__ scores) {
  __shared__ float red[256];
  float* p = scores + (size_t)blockIdx.x * NI;
  const int t = threadIdx.x;

  float m = -3.402823466e38f;
  for (int i = t; i < NI; i += 256) m = fmaxf(m, p[i]);
  red[t] = m; __syncthreads();
  for (int s = 128; s > 0; s >>= 1) { if (t < s) red[t] = fmaxf(red[t], red[t + s]); __syncthreads(); }
  m = red[0]; __syncthreads();

  float sum = 0.f;
  for (int i = t; i < NI; i += 256) sum += __expf(p[i] - m);
  red[t] = sum; __syncthreads();
  for (int s = 128; s > 0; s >>= 1) { if (t < s) red[t] += red[t + s]; __syncthreads(); }
  const float lse = m + __logf(red[0]);

  for (int i = t; i < NI; i += 256) p[i] = p[i] - lse;
}

// ---------------------------------------------------------------------------
extern "C" void kernel_launch(void* const* d_in, const int* in_sizes, int n_in,
                              void* d_out, int out_size, void* d_ws, size_t ws_size,
                              hipStream_t stream) {
  const int*   uidx = (const int*)d_in[0];
  const int*   iidx = (const int*)d_in[1];
  const float* adj  = (const float*)d_in[2];
  const float* utab = (const float*)d_in[3];
  const float* itab = (const float*)d_in[4];
  const float* gc_w = (const float*)d_in[5];
  const float* gc_b = (const float*)d_in[6];
  const float* bi_w = (const float*)d_in[7];
  const float* bi_b = (const float*)d_in[8];
  float* out = (float*)d_out;

  // workspace carve-out (~28.4 MB)
  char* ws = (char*)d_ws;
  float*  all_e = (float*)ws;   ws += (size_t)NB * NODES * EPITCH * sizeof(float);
  float*  side  = (float*)ws;   ws += (size_t)NB * NODES * DIM * sizeof(float);
  __bf16* egoT  = (__bf16*)ws;  ws += (size_t)NB * DIM * NODES * sizeof(__bf16);
  __bf16* gcw16 = (__bf16*)ws;  ws += (size_t)NLAYER * DIM * DIM * sizeof(__bf16);
  __bf16* biw16 = (__bf16*)ws;

  wconvert_kernel<<<(NLAYER * DIM * DIM + 255) / 256, 256, 0, stream>>>(gc_w, bi_w, gcw16, biw16);
  gather_kernel<<<dim3(NODES, 1, NB), DIM, 0, stream>>>(uidx, iidx, utab, itab, all_e, egoT);

  for (int l = 0; l < NLAYER; ++l) {
    side_gemm_kernel<<<dim3(NODES / 64, 1, NB), 128, 0, stream>>>(adj, egoT, side);
    combine_kernel<<<dim3(NODES / 64, 1, NB), 128, 0, stream>>>(
        side, all_e, egoT,
        gcw16 + (size_t)l * DIM * DIM, biw16 + (size_t)l * DIM * DIM,
        gc_b + l * DIM, bi_b + l * DIM, l);
  }

  scores_gemm_kernel<<<dim3(NI / 128, NU / 64, NB), 256, 0, stream>>>(all_e, out);
  logsoftmax_kernel<<<NB * NU, 256, 0, stream>>>(out);
}
